// LSTMSubNet_35347580846480
// MI455X (gfx1250) — matrix operand order repaired and checked by reading, not compiled, not run on hardware
//
#include <hip/hip_runtime.h>
#include <hip/hip_bf16.h>

// Problem constants (from reference): B=128, T=1024, I=300, H=128, O=64
#define B_   128
#define T_   1024
#define I_   300
#define IP   320       // K padded to multiple of 32 for WMMA
#define H_   128
#define G4   512       // 4*H
#define O_   64
#define BT   (B_ * T_) // 131072

typedef __attribute__((ext_vector_type(4)))  float  v4f;
typedef __attribute__((ext_vector_type(8)))  float  v8f;
typedef __attribute__((ext_vector_type(8)))  __bf16 v8bf;
typedef __attribute__((ext_vector_type(16))) __bf16 v16bf;

static __device__ __forceinline__ v8f wmma_bf16(v16bf a, v16bf b, v8f c) {
  // D = A(16x32 bf16) x B(32x16 bf16) + C(16x16 f32)
  return __builtin_amdgcn_wmma_f32_16x16x32_bf16(false, a, false, b, (short)0, c,
                                                 false, false);
}

static __device__ __forceinline__ v16bf cat8(v8bf lo, v8bf hi) {
  return __builtin_shufflevector(lo, hi, 0, 1, 2, 3, 4, 5, 6, 7, 8, 9, 10, 11, 12,
                                 13, 14, 15);
}

static __device__ __forceinline__ float sigmoidf_(float x) {
  return 1.0f / (1.0f + __expf(-x));
}

// ---------------------------------------------------------------------------
// Prep: fp32 weights -> bf16 (W_ih zero-padded K 300 -> 320)
// ---------------------------------------------------------------------------
__global__ __launch_bounds__(256) void k_prep(const float* __restrict__ Wih,
                                              const float* __restrict__ Whh,
                                              const float* __restrict__ W1,
                                              __bf16* __restrict__ Wih_b,
                                              __bf16* __restrict__ Whh_b,
                                              __bf16* __restrict__ W1_b) {
  const int NIH = G4 * IP;   // 163840
  const int NHH = G4 * H_;   // 65536
  const int NW1 = O_ * H_;   // 8192
  int idx = blockIdx.x * 256 + threadIdx.x;
  if (idx < NIH) {
    int n = idx / IP, k = idx % IP;
    Wih_b[idx] = (k < I_) ? (__bf16)Wih[n * I_ + k] : (__bf16)0.0f;
  } else if (idx < NIH + NHH) {
    int j = idx - NIH;
    Whh_b[j] = (__bf16)Whh[j];
  } else if (idx < NIH + NHH + NW1) {
    int j = idx - NIH - NHH;
    W1_b[j] = (__bf16)W1[j];
  }
}

// ---------------------------------------------------------------------------
// Input GEMM: xg[bt, g] = x[bt, :] . W_ih[g, :] + b_ih[g] + b_hh[g]
// Block = 16 rows x 512 cols, 8 waves; wave w covers gate cols [64w, 64w+64).
// ---------------------------------------------------------------------------
__global__ __launch_bounds__(256) void k_in_gemm(const float* __restrict__ x,
                                                 const __bf16* __restrict__ Wih,
                                                 const float* __restrict__ b_ih,
                                                 const float* __restrict__ b_hh,
                                                 float* __restrict__ xg) {
  const int lane = threadIdx.x & 31;
  const int wave = threadIdx.x >> 5;
  const int hh = lane >> 4;   // 0: lanes 0-15, 1: lanes 16-31
  const int ln = lane & 15;
  const size_t row0 = (size_t)blockIdx.x * 16;
  const int n0w = wave * 64;

  v8f acc[4] = {};
#pragma unroll
  for (int kc = 0; kc < 10; ++kc) {
    const int k0 = kc * 32;
    // A fragment: row = row0+ln, chunks at k0+8*hh and k0+8*hh+16 (fp32->bf16)
    v16bf a;
    {
      const size_t rbase = (row0 + ln) * (size_t)I_;
      const int ka = k0 + 8 * hh;
      if (kc < 9) {  // fully in-range (max idx 287 < 300), aligned vector loads
        const v4f c0 = *(const v4f*)(x + rbase + ka);
        const v4f c1 = *(const v4f*)(x + rbase + ka + 4);
        const v4f c2 = *(const v4f*)(x + rbase + ka + 16);
        const v4f c3 = *(const v4f*)(x + rbase + ka + 20);
#pragma unroll
        for (int j = 0; j < 4; ++j) {
          a[j]      = (__bf16)c0[j];
          a[4 + j]  = (__bf16)c1[j];
          a[8 + j]  = (__bf16)c2[j];
          a[12 + j] = (__bf16)c3[j];
        }
      } else {  // K tail 288..299, zero-pad to 320
#pragma unroll
        for (int j = 0; j < 8; ++j) {
          const int k1 = ka + j, k2 = ka + 16 + j;
          a[j]     = (k1 < I_) ? (__bf16)x[rbase + k1] : (__bf16)0.0f;
          a[8 + j] = (k2 < I_) ? (__bf16)x[rbase + k2] : (__bf16)0.0f;
        }
      }
    }
    // B fragments: col n -> W_ih row n, chunks at k0+8*hh and k0+8*hh+16
#pragma unroll
    for (int nt = 0; nt < 4; ++nt) {
      const __bf16* bp = Wih + (size_t)(n0w + nt * 16 + ln) * IP + k0 + 8 * hh;
      const v16bf b = cat8(*(const v8bf*)bp, *(const v8bf*)(bp + 16));
      acc[nt] = wmma_bf16(a, b, acc[nt]);
    }
  }
#pragma unroll
  for (int nt = 0; nt < 4; ++nt) {
    const int col = n0w + nt * 16 + ln;
    const float bias = b_ih[col] + b_hh[col];
#pragma unroll
    for (int r = 0; r < 8; ++r) {
      const size_t row = row0 + r + 8 * hh;
      xg[row * G4 + col] = acc[nt][r] + bias;
    }
  }
}

// ---------------------------------------------------------------------------
// Recurrent kernel: 8 WGs, each owns 16 batch rows for all 1024 steps.
// W_hh bf16 fragments resident in VGPRs; h in LDS (bf16); c in registers.
// ---------------------------------------------------------------------------
__global__ __launch_bounds__(256) void k_lstm(const float* __restrict__ xg,
                                              const __bf16* __restrict__ Whh,
                                              __bf16* __restrict__ hseq) {
  __shared__ float gates[16][G4];   // 32 KB: activated i|f|g|o
  __shared__ __bf16 hsh[16][H_];    // 4 KB: current h (bf16)

  const int tid = threadIdx.x;
  const int lane = tid & 31, wave = tid >> 5;
  const int hh = lane >> 4, ln = lane & 15;
  const int b0 = blockIdx.x * 16;

  // Resident W_hh B fragments: wave covers gate cols [64*wave, 64*wave+64)
  v16bf Bh[4][4];
#pragma unroll
  for (int nt = 0; nt < 4; ++nt)
#pragma unroll
    for (int kc = 0; kc < 4; ++kc)
      Bh[nt][kc] = *(const v16bf*)(Whh + (size_t)(wave * 64 + nt * 16 + ln) * H_ +
                                   kc * 32 + 16 * hh);

  // Per-thread cell state: row = tid>>4, cols [8*(tid&15), +8)
  const int crow = tid >> 4;
  const int ccol = (tid & 15) * 8;
  float cst[8];
  v8bf hz;
#pragma unroll
  for (int j = 0; j < 8; ++j) { cst[j] = 0.0f; hz[j] = (__bf16)0.0f; }
  *(v8bf*)&hsh[crow][ccol] = hz;  // h0 = 0
  __syncthreads();

  for (int t = 0; t < T_; ++t) {
    // Accumulators start at precomputed input gates
    v8f acc[4];
#pragma unroll
    for (int nt = 0; nt < 4; ++nt) {
      const int col = wave * 64 + nt * 16 + ln;
#pragma unroll
      for (int r = 0; r < 8; ++r) {
        const size_t bt = (size_t)(b0 + r + 8 * hh) * T_ + t;
        acc[nt][r] = xg[bt * G4 + col];
      }
    }
    // gates += h @ W_hh^T  (K = 128 -> 4 WMMA k-chunks)
#pragma unroll
    for (int kc = 0; kc < 4; ++kc) {
      const int ka = kc * 32 + 16 * hh;
      const v8bf lo = *(const v8bf*)&hsh[ln][ka];
      const v8bf hi = *(const v8bf*)&hsh[ln][ka + 8];
      const v16bf a = cat8(lo, hi);
#pragma unroll
      for (int nt = 0; nt < 4; ++nt) acc[nt] = wmma_bf16(a, Bh[nt][kc], acc[nt]);
    }
    // Activations -> LDS (i,f,o: sigmoid; g: tanh)
#pragma unroll
    for (int nt = 0; nt < 4; ++nt) {
      const int col = wave * 64 + nt * 16 + ln;
      const bool is_g = ((col >> 7) == 2);
#pragma unroll
      for (int r = 0; r < 8; ++r) {
        const float v = acc[nt][r];
        gates[r + 8 * hh][col] = is_g ? tanhf(v) : sigmoidf_(v);
      }
    }
    __syncthreads();
    // Cell/hidden update (8 contiguous h columns per thread)
    v8bf hv;
#pragma unroll
    for (int j = 0; j < 8; ++j) {
      const int col = ccol + j;
      const float ig = gates[crow][col];
      const float fg = gates[crow][H_ + col];
      const float gg = gates[crow][2 * H_ + col];
      const float og = gates[crow][3 * H_ + col];
      const float c = fg * cst[j] + ig * gg;
      cst[j] = c;
      hv[j] = (__bf16)(og * tanhf(c));
    }
    *(v8bf*)&hsh[crow][ccol] = hv;
    const size_t bt = (size_t)(b0 + crow) * T_ + t;
    *(v8bf*)(hseq + bt * H_ + ccol) = hv;
    __syncthreads();
  }
}

// ---------------------------------------------------------------------------
// Output projection: y[bt, o] = hseq[bt, :] . W1[o, :] + b1[o]
// Wave computes one 16-row strip x all 64 cols (4 N-tiles), K = 128.
// ---------------------------------------------------------------------------
__global__ __launch_bounds__(256) void k_proj(const __bf16* __restrict__ hseq,
                                              const __bf16* __restrict__ W1b,
                                              const float* __restrict__ b1,
                                              float* __restrict__ y) {
  const int lane = threadIdx.x & 31, wave = threadIdx.x >> 5;
  const int hh = lane >> 4, ln = lane & 15;
  const size_t row0 = ((size_t)blockIdx.x * 8 + wave) * 16;

  v8f acc[4] = {};
#pragma unroll
  for (int kc = 0; kc < 4; ++kc) {
    const int ka = kc * 32 + 16 * hh;
    const __bf16* ap = hseq + (row0 + ln) * H_ + ka;
    const v16bf a = cat8(*(const v8bf*)ap, *(const v8bf*)(ap + 8));
#pragma unroll
    for (int nt = 0; nt < 4; ++nt) {
      const v16bf b =
          *(const v16bf*)(W1b + (size_t)(nt * 16 + ln) * H_ + kc * 32 + 16 * hh);
      acc[nt] = wmma_bf16(a, b, acc[nt]);
    }
  }
#pragma unroll
  for (int nt = 0; nt < 4; ++nt) {
    const int col = nt * 16 + ln;
    const float bias = b1[col];
#pragma unroll
    for (int r = 0; r < 8; ++r)
      y[(row0 + r + 8 * hh) * O_ + col] = acc[nt][r] + bias;
  }
}

// ---------------------------------------------------------------------------
extern "C" void kernel_launch(void* const* d_in, const int* in_sizes, int n_in,
                              void* d_out, int out_size, void* d_ws, size_t ws_size,
                              hipStream_t stream) {
  const float* x    = (const float*)d_in[0];
  const float* Wih  = (const float*)d_in[1];
  const float* Whh  = (const float*)d_in[2];
  const float* b_ih = (const float*)d_in[3];
  const float* b_hh = (const float*)d_in[4];
  const float* W1   = (const float*)d_in[5];
  const float* b1   = (const float*)d_in[6];
  float* y = (float*)d_out;

  // Workspace layout (256B aligned): ~302 MB total
  char* ws = (char*)d_ws;
  size_t off = 0;
  auto take = [&](size_t bytes) -> void* {
    void* p = ws + off;
    off += (bytes + 255) & ~(size_t)255;
    return p;
  };
  __bf16* Wih_b = (__bf16*)take((size_t)G4 * IP * 2);   // 320 KB
  __bf16* Whh_b = (__bf16*)take((size_t)G4 * H_ * 2);   // 128 KB
  __bf16* W1_b  = (__bf16*)take((size_t)O_ * H_ * 2);   // 16 KB
  float*  xg    = (float*)take((size_t)BT * G4 * 4);    // 256 MB
  __bf16* hseq  = (__bf16*)take((size_t)BT * H_ * 2);   // 32 MB

  const int prep_n = G4 * IP + G4 * H_ + O_ * H_;
  k_prep<<<(prep_n + 255) / 256, 256, 0, stream>>>(Wih, Whh, W1, Wih_b, Whh_b, W1_b);
  k_in_gemm<<<BT / 16, 256, 0, stream>>>(x, Wih_b, b_ih, b_hh, xg);
  k_lstm<<<B_ / 16, 256, 0, stream>>>(xg, Whh_b, hseq);
  k_proj<<<BT / 128, 256, 0, stream>>>(hseq, W1_b, b1, y);
}
